// GGNNMeanResidual_78151224918835
// MI455X (gfx1250) — matrix operand order repaired and checked by reading, not compiled, not run on hardware
//
#include <hip/hip_runtime.h>
#include <hip/hip_bf16.h>

typedef __bf16 bf16_t;
typedef __attribute__((ext_vector_type(16))) __bf16 v16bf;
typedef __attribute__((ext_vector_type(8)))  float  v8f;

#define NN     65536      // nodes
#define NE     524288     // edges
#define DD     128
#define NET    4
#define NSTEPS 8
#define NG     64
#define NHID   256

// ---------------- WMMA helpers ----------------

__device__ __forceinline__ v8f wmma_bf16(v16bf a, v16bf b, v8f c) {
  return __builtin_amdgcn_wmma_f32_16x16x32_bf16(false, a, false, b, (short)0, c,
                                                 false, false);
}

// A-matrix fragment (16 rows x K-chunk of 32), bf16 row-major source, ld in elems.
// Lane r in [0,16): row r. hi half selects K sub-bands {0-7,16-23} vs {8-15,24-31}.
__device__ __forceinline__ v16bf load_fragA(const bf16_t* __restrict__ base, int ld,
                                            int row16, int k0, int lane) {
  int r = lane & 15, hi = lane >> 4;
  const bf16_t* p = base + (size_t)(row16 + r) * ld + k0 + hi * 8;
  union { v16bf v; uint4 u[2]; } f;
  f.u[0] = *reinterpret_cast<const uint4*>(p);        // K = k0+hi*8 .. +7
  f.u[1] = *reinterpret_cast<const uint4*>(p + 16);   // K = k0+16+hi*8 .. +7
  return f.v;
}

// B-matrix fragment: C = A * W^T, W row-major [ncols][K]; lane n holds W row
// (row16+n); hi half selects K sub-bands {0-7,16-23} vs {8-15,24-31}.
__device__ __forceinline__ v16bf load_fragB(const bf16_t* __restrict__ base, int ld,
                                            int row16, int k0, int lane) {
  int n = lane & 15, hi = lane >> 4;
  const bf16_t* p = base + (size_t)(row16 + n) * ld + k0 + hi * 8;
  union { v16bf v; uint4 u[2]; } f;
  f.u[0] = *reinterpret_cast<const uint4*>(p);
  f.u[1] = *reinterpret_cast<const uint4*>(p + 16);
  return f.v;
}

__device__ __forceinline__ void atomic_addf(float* p, float v) {
  __hip_atomic_fetch_add(p, v, __ATOMIC_RELAXED, __HIP_MEMORY_SCOPE_AGENT);
}

// CDNA5 packed 2xbf16 atomic add (GLOBAL_ATOMIC_PK_ADD_BF16, no return ->
// tracked on STOREcnt; s_endpgm's implicit wait covers completion).
__device__ __forceinline__ void atomic_pk_add_bf16(unsigned long long addr,
                                                   unsigned int data) {
  asm volatile("global_atomic_pk_add_bf16 %0, %1, off"
               :
               : "v"(addr), "v"(data)
               : "memory");
}

// ---------------- kernels ----------------

__global__ void k_init(const float* __restrict__ f, float* __restrict__ hf,
                       bf16_t* __restrict__ hb) {
  size_t i = (size_t)blockIdx.x * 256 + threadIdx.x;
  float v = f[i];
  hf[i] = v;
  hb[i] = (bf16_t)v;
}

__global__ void k_wcvt(const float* __restrict__ We, const float* __restrict__ wih,
                       const float* __restrict__ whh, bf16_t* __restrict__ Web,
                       bf16_t* __restrict__ wihb, bf16_t* __restrict__ whhb) {
  int i = blockIdx.x * 256 + threadIdx.x;
  if (i < NET * DD * DD) Web[i] = (bf16_t)We[i];
  if (i < 3 * DD * DD) { wihb[i] = (bf16_t)wih[i]; whhb[i] = (bf16_t)whh[i]; }
}

__global__ void k_zero(float* __restrict__ p, int n) {
  int i = blockIdx.x * 256 + threadIdx.x;
  if (i < n) p[i] = 0.f;
}

// T[e][n][:] = bf16( h[n][:] @ W_e^T ).  8 waves/block, one 16x16 tile per wave.
__global__ void k_transform(const bf16_t* __restrict__ h, const bf16_t* __restrict__ Wet,
                            bf16_t* __restrict__ T) {
  int lane = threadIdx.x & 31;
  int gw   = blockIdx.x * 8 + (threadIdx.x >> 5);
  int mt = gw >> 5;            // node tile (0..4095)
  int r  = gw & 31;
  int e  = r >> 3;             // etype
  int ct = r & 7;              // 16-col tile
  v8f acc = {};
#pragma unroll
  for (int kc = 0; kc < 4; ++kc) {
    v16bf fa = load_fragA(h, DD, mt * 16, kc * 32, lane);
    v16bf fb = load_fragB(Wet + (size_t)e * DD * DD, DD, ct * 16, kc * 32, lane);
    acc = wmma_bf16(fa, fb, acc);
  }
  int n = lane & 15, hi = lane >> 4;
  bf16_t* out = T + (((size_t)e * NN + mt * 16) * DD) + ct * 16 + n;
#pragma unroll
  for (int g = 0; g < 8; ++g) out[(size_t)(g + 8 * hi) * DD] = (bf16_t)acc[g];
}

// a[dst] += T[etype][src] + b[etype] using packed bf16 atomics.
// 4 edges per 256-thread block; 64 lanes per edge, each lane owns a dim pair.
__global__ void k_scatter(const int* __restrict__ src, const int* __restrict__ dst,
                          const int* __restrict__ et, const bf16_t* __restrict__ T,
                          const float* __restrict__ eb, bf16_t* __restrict__ a) {
  int e = blockIdx.x * 4 + (threadIdx.x >> 6);
  int j = threadIdx.x & 63;                    // bf16 pair index (dims 2j, 2j+1)
  int t = et[e];
  const unsigned int* trow = reinterpret_cast<const unsigned int*>(
      T + ((size_t)t * NN + src[e]) * DD);
  union { unsigned int u; __bf16 b[2]; } in, outp;
  in.u = trow[j];
  outp.b[0] = (bf16_t)((float)in.b[0] + eb[t * DD + 2 * j]);
  outp.b[1] = (bf16_t)((float)in.b[1] + eb[t * DD + 2 * j + 1]);
  const unsigned int* arow = reinterpret_cast<const unsigned int*>(a) +
                             (size_t)dst[e] * 64 + j;
  atomic_pk_add_bf16((unsigned long long)arow, outp.u);
}

// Fused GRU: gi = a@w_ih^T, gh = h@w_hh^T (48 WMMA tiles -> LDS), then gates.
// Updates h_f32 / h_bf16 in place (block owns rows m0..m0+15).
__global__ void k_gru(const bf16_t* __restrict__ a, float* __restrict__ hf,
                      bf16_t* __restrict__ hb, const bf16_t* __restrict__ wih,
                      const bf16_t* __restrict__ whh, const float* __restrict__ bih,
                      const float* __restrict__ bhh) {
  __shared__ float gi[16 * 384];
  __shared__ float gh[16 * 384];
  int lane = threadIdx.x & 31, wave = threadIdx.x >> 5;
  int m0 = blockIdx.x * 16;

  v16bf fa[4], fh[4];
#pragma unroll
  for (int kc = 0; kc < 4; ++kc) {
    fa[kc] = load_fragA(a, DD, m0, kc * 32, lane);
    fh[kc] = load_fragA(hb, DD, m0, kc * 32, lane);
  }
#pragma unroll
  for (int i = 0; i < 6; ++i) {
    int tt = wave + i * 8;                 // 0..47
    bool is_gi = tt < 24;
    int ct = is_gi ? tt : tt - 24;
    const bf16_t* w = is_gi ? wih : whh;
    v8f acc = {};
#pragma unroll
    for (int kc = 0; kc < 4; ++kc) {
      v16bf fb = load_fragB(w, DD, ct * 16, kc * 32, lane);
      v16bf av;
      if (is_gi) av = fa[kc]; else av = fh[kc];
      acc = wmma_bf16(av, fb, acc);
    }
    float* dstp = (is_gi ? gi : gh) + ct * 16 + (lane & 15);
    int hi = lane >> 4;
#pragma unroll
    for (int g = 0; g < 8; ++g) dstp[(g + 8 * hi) * 384] = acc[g];
  }
  __syncthreads();

#pragma unroll
  for (int i = 0; i < 8; ++i) {
    int idx = threadIdx.x + i * 256;       // 0..2047
    int m = idx >> 7, d2 = idx & 127;
    float ir = gi[m * 384 + d2]        + bih[d2];
    float iz = gi[m * 384 + 128 + d2]  + bih[128 + d2];
    float in_= gi[m * 384 + 256 + d2]  + bih[256 + d2];
    float hr = gh[m * 384 + d2]        + bhh[d2];
    float hz = gh[m * 384 + 128 + d2]  + bhh[128 + d2];
    float hn = gh[m * 384 + 256 + d2]  + bhh[256 + d2];
    float rg = 1.f / (1.f + __expf(-(ir + hr)));
    float zg = 1.f / (1.f + __expf(-(iz + hz)));
    float ng = tanhf(in_ + rg * hn);
    size_t off = (size_t)(m0 + m) * DD + d2;
    float hold = hf[off];
    float hnew = (1.f - zg) * ng + zg * hold;
    hf[off] = hnew;
    hb[off] = (bf16_t)hnew;
  }
}

// Sorted-run mean-pool: block handles 128 consecutive nodes, thread d owns one
// of the 256 concat dims; flush an atomic only when graph id changes.
__global__ void k_pool(const float* __restrict__ h, const float* __restrict__ feat,
                       const int* __restrict__ gid, float* __restrict__ sums,
                       float* __restrict__ cnts) {
  int n0 = blockIdx.x * 128;
  int d2 = threadIdx.x;
  float s = 0.f, c = 0.f;
  int cur = gid[n0];
  for (int i = 0; i < 128; ++i) {
    int n = n0 + i;
    int g = gid[n];
    if (g != cur) {
      atomic_addf(&sums[cur * 256 + d2], s);
      if (d2 == 0) atomic_addf(&cnts[cur], c);
      s = 0.f; c = 0.f; cur = g;
    }
    float v = (d2 < 128) ? h[(size_t)n * DD + d2] : feat[(size_t)n * DD + (d2 - 128)];
    s += v; c += 1.f;
  }
  atomic_addf(&sums[cur * 256 + d2], s);
  if (d2 == 0) atomic_addf(&cnts[cur], c);
}

// pooled -> relu(W1) -> W2 -> sigmoid.  64x256 problem: one block.
__global__ void k_cls(const float* __restrict__ sums, const float* __restrict__ cnts,
                      const float* __restrict__ W1, const float* __restrict__ b1,
                      const float* __restrict__ W2, const float* __restrict__ b2,
                      float* __restrict__ hc, float* __restrict__ out) {
  __shared__ float pooled[NG * 256];
  for (int i = threadIdx.x; i < NG * 256; i += 256) {
    float c = cnts[i >> 8];
    pooled[i] = sums[i] / (c < 1.f ? 1.f : c);
  }
  __syncthreads();
  for (int o = threadIdx.x; o < NG * NHID; o += 256) {
    int g = o >> 8, j = o & 255;
    float acc = b1[j];
    const float* w = W1 + (size_t)j * 256;
    const float* p = pooled + g * 256;
    for (int k = 0; k < 256; ++k) acc += p[k] * w[k];
    hc[o] = acc > 0.f ? acc : 0.f;
  }
  __syncthreads();
  if (threadIdx.x < NG) {
    int g = threadIdx.x;
    float acc = b2[0];
    const float* p = hc + (size_t)g * NHID;
    for (int k = 0; k < NHID; ++k) acc += p[k] * W2[k];
    out[g] = 1.f / (1.f + __expf(-acc));
  }
}

// ---------------- launch ----------------

extern "C" void kernel_launch(void* const* d_in, const int* in_sizes, int n_in,
                              void* d_out, int out_size, void* d_ws, size_t ws_size,
                              hipStream_t stream) {
  const float* features = (const float*)d_in[0];
  const int*   src      = (const int*)d_in[1];
  const int*   dst      = (const int*)d_in[2];
  const int*   ety      = (const int*)d_in[3];
  const int*   gid      = (const int*)d_in[4];
  const float* etW      = (const float*)d_in[5];
  const float* etb      = (const float*)d_in[6];
  const float* wih      = (const float*)d_in[7];
  const float* whh      = (const float*)d_in[8];
  const float* bih      = (const float*)d_in[9];
  const float* bhh      = (const float*)d_in[10];
  const float* W1       = (const float*)d_in[11];
  const float* b1       = (const float*)d_in[12];
  const float* W2       = (const float*)d_in[13];
  const float* b2       = (const float*)d_in[14];
  float* out = (float*)d_out;

  char* ws = (char*)d_ws;
  size_t off = 0;
  auto alloc = [&](size_t bytes) -> void* {
    void* p = ws + off;
    off += (bytes + 255) & ~(size_t)255;
    return p;
  };
  float*  h_f  = (float*) alloc((size_t)NN * DD * 4);        // 32 MB
  bf16_t* h_b  = (bf16_t*)alloc((size_t)NN * DD * 2);        // 16 MB
  bf16_t* a_b  = (bf16_t*)alloc((size_t)NN * DD * 2);        // 16 MB (bf16 accum)
  bf16_t* T_b  = (bf16_t*)alloc((size_t)NET * NN * DD * 2);  // 64 MB
  bf16_t* Web  = (bf16_t*)alloc((size_t)NET * DD * DD * 2);
  bf16_t* wihb = (bf16_t*)alloc((size_t)3 * DD * DD * 2);
  bf16_t* whhb = (bf16_t*)alloc((size_t)3 * DD * DD * 2);
  float*  sums = (float*) alloc((size_t)NG * 256 * 4 + NG * 4);  // sums + cnts
  float*  cnts = sums + NG * 256;
  float*  hc   = (float*) alloc((size_t)NG * NHID * 4);

  k_init<<<(NN * DD) / 256, 256, 0, stream>>>(features, h_f, h_b);
  k_wcvt<<<(NET * DD * DD + 255) / 256, 256, 0, stream>>>(etW, wih, whh, Web, wihb, whhb);

  for (int s = 0; s < NSTEPS; ++s) {
    k_transform<<<(NN / 16) * 32 / 8, 256, 0, stream>>>(h_b, Web, T_b);
    // zero bf16 accumulator (NN*DD bf16 == NN*DD/2 floats; 0x0 == +0.0bf16 pair)
    k_zero<<<(NN * DD / 2 + 255) / 256, 256, 0, stream>>>((float*)a_b, NN * DD / 2);
    k_scatter<<<NE / 4, 256, 0, stream>>>(src, dst, ety, T_b, etb, a_b);
    k_gru<<<NN / 16, 256, 0, stream>>>(a_b, h_f, h_b, wihb, whhb, bih, bhh);
  }

  k_zero<<<(NG * 256 + NG + 255) / 256, 256, 0, stream>>>(sums, NG * 256 + NG);
  k_pool<<<NN / 128, 256, 0, stream>>>(h_f, features, gid, sums, cnts);
  k_cls<<<1, 256, 0, stream>>>(sums, cnts, W1, b1, W2, b2, hc, out);
}
